// CoodinatePredictionModel_91336774516992
// MI455X (gfx1250) — matrix operand order repaired and checked by reading, not compiled, not run on hardware
//
#include <hip/hip_runtime.h>

typedef __attribute__((ext_vector_type(16))) _Float16 v16h;
typedef __attribute__((ext_vector_type(8)))  _Float16 v8h;
typedef __attribute__((ext_vector_type(8)))  float    v8f;
typedef __attribute__((ext_vector_type(4)))  float    v4f;

#define T_STEPS   200
#define DEC_STEPS 30
#define B_TOT     4096
#define IN_DIM    4
#define OUT_DIM   4
#define H_DIM     100
#define G_DIM     400      // 4*H
#define BT        32       // batch rows per workgroup
#define NTILES    25       // 400 / 16
#define MTILES    2        // 32 / 16
#define KSTEPS    4        // Kpad 128 / 32
#define NWAVES    8
#define THREADS   256

// ---- LDS layout (element counts) ----
#define WFRAG_HALFS   (KSTEPS * NTILES * 512)   // 51200 halfs = 100 KB, B-fragments
#define HFRAG_STRIDE  136                       // 128 + 8 pad (bank decorrelation)
#define HFRAG_HALFS   (BT * HFRAG_STRIDE)
#define GT_STRIDE     36                        // transposed gates [n][b], 32 + 4 pad
#define GATES_FLOATS  (G_DIM * GT_STRIDE)       // 14400 floats
#define HC_STRIDE     104

#define SMEM_BYTES (WFRAG_HALFS*2 + HFRAG_HALFS*2 + \
                    (GATES_FLOATS + 2*BT*HC_STRIDE + G_DIM*IN_DIM + G_DIM + \
                     BT*IN_DIM + OUT_DIM*H_DIM + OUT_DIM) * 4)

__device__ __forceinline__ v16h combine16(v8h lo, v8h hi) {
  v16h r;
#pragma unroll
  for (int i = 0; i < 8; ++i) { r[i] = lo[i]; r[i + 8] = hi[i]; }
  return r;
}

__device__ __forceinline__ float sigf(float x) {
  return 1.0f / (1.0f + __expf(-x));
}
__device__ __forceinline__ float tanh_fast(float x) {
  x = fminf(fmaxf(x, -15.0f), 15.0f);
  float e = __expf(2.0f * x);
  return (e - 1.0f) / (e + 1.0f);
}

// Pack W (G_DIM x H_DIM, row-major, gates = h @ W^T) into WMMA B-fragment layout:
// frag[(ks*NTILES+nt)*512 + lane*16 + t] holds B[k][n] with
//   n = nt*16 + (lane&15),  k = ks*32 + (lane>=16 ? 16 : 0) + t   (zero-pad k>=100)
__device__ __forceinline__ void build_wfrag(_Float16* wfrag, const float* W, int tid) {
  for (int idx = tid; idx < WFRAG_HALFS; idx += THREADS) {
    int t    = idx & 15;
    int lane = (idx >> 4) & 31;
    int tile = idx >> 9;
    int ks   = tile / NTILES;
    int nt   = tile - ks * NTILES;
    int n    = nt * 16 + (lane & 15);
    int k    = ks * 32 + ((lane >= 16) ? 16 : 0) + t;
    float w  = (k < H_DIM) ? W[n * H_DIM + k] : 0.0f;
    wfrag[idx] = (_Float16)w;
  }
}

// One recurrent GEMM step: gatesT[400][32] = (h @ Whh^T)^T, f32 accum.
// A-fragments hoisted (invariant across n-tiles); B-fragment reused by both m-tiles.
// Transposed gate staging: lane's 8 accumulator values are contiguous -> b128 stores.
__device__ __forceinline__ void gemm_step(const _Float16* __restrict__ wfrag,
                                          const _Float16* __restrict__ hfrag,
                                          float* __restrict__ gatesT, int tid) {
  const int wave = __builtin_amdgcn_readfirstlane(tid >> 5);  // scalar wave id
  const int lane = tid & 31;
  const int l15  = lane & 15;     // A: row-in-tile / C,D: col-in-tile
  const int hi   = lane >> 4;     // upper half-lane group

  // Hoist all A fragments: a[mt][ks], invariant over the nt loop.
  v16h a[MTILES][KSTEPS];
#pragma unroll
  for (int mt = 0; mt < MTILES; ++mt) {
#pragma unroll
    for (int ks = 0; ks < KSTEPS; ++ks) {
      const _Float16* ap =
          hfrag + (mt * 16 + l15) * HFRAG_STRIDE + ks * 32 + (hi << 4);
      a[mt][ks] = combine16(*(const v8h*)ap, *(const v8h*)(ap + 8));
    }
  }

  for (int nt = wave; nt < NTILES; nt += NWAVES) {
    // Load all 4 B fragments first (in-order LDS returns -> partial waits)
    v16h b[KSTEPS];
#pragma unroll
    for (int ks = 0; ks < KSTEPS; ++ks) {
      const _Float16* bp = wfrag + (ks * NTILES + nt) * 512 + lane * 16;
      b[ks] = combine16(*(const v8h*)bp, *(const v8h*)(bp + 8));
    }
    v8f acc0 = {}, acc1 = {};
#pragma unroll
    for (int ks = 0; ks < KSTEPS; ++ks) {
      acc0 = __builtin_amdgcn_wmma_f32_16x16x32_f16(
          false, a[0][ks], false, b[ks], (short)0, acc0, false, false);
      acc1 = __builtin_amdgcn_wmma_f32_16x16x32_f16(
          false, a[1][ks], false, b[ks], (short)0, acc1, false, false);
    }
    // C/D layout: VGPR v -> row (mt*16 + 8*hi + v), col n = nt*16 + l15.
    // gatesT[n][b] with stride 36 -> lane's 8 values contiguous: 2x b128 per acc.
    float* gp = gatesT + (nt * 16 + l15) * GT_STRIDE + (hi << 3);
    v4f s;
    s[0] = acc0[0]; s[1] = acc0[1]; s[2] = acc0[2]; s[3] = acc0[3];
    *(v4f*)(gp + 0) = s;
    s[0] = acc0[4]; s[1] = acc0[5]; s[2] = acc0[6]; s[3] = acc0[7];
    *(v4f*)(gp + 4) = s;
    s[0] = acc1[0]; s[1] = acc1[1]; s[2] = acc1[2]; s[3] = acc1[3];
    *(v4f*)(gp + 16) = s;
    s[0] = acc1[4]; s[1] = acc1[5]; s[2] = acc1[6]; s[3] = acc1[7];
    *(v4f*)(gp + 20) = s;
  }
}

// Elementwise LSTM cell: fuse x@Wih^T + biases into gates, update c/h.
// b fixed per thread (tid>>3); j strided by 8 -> no integer division.
__device__ __forceinline__ void cell_step(const float* __restrict__ gatesT,
                                          const float* __restrict__ wih,
                                          const float* __restrict__ bias,
                                          const float* __restrict__ xl,
                                          float* __restrict__ cbuf,
                                          float* __restrict__ hf32,
                                          _Float16* __restrict__ hfrag, int tid) {
  const int b  = tid >> 3;      // 0..31
  const int j0 = tid & 7;
  const float x0 = xl[b * 4 + 0], x1 = xl[b * 4 + 1];
  const float x2 = xl[b * 4 + 2], x3 = xl[b * 4 + 3];
  for (int j = j0; j < H_DIM; j += 8) {
    float g[4];
#pragma unroll
    for (int q = 0; q < 4; ++q) {
      const int n = q * H_DIM + j;
      const float* wr = wih + n * 4;
      g[q] = gatesT[n * GT_STRIDE + b] + bias[n] +
             x0 * wr[0] + x1 * wr[1] + x2 * wr[2] + x3 * wr[3];
    }
    const float c  = cbuf[b * HC_STRIDE + j];
    const float cn = sigf(g[1]) * c + sigf(g[0]) * tanh_fast(g[2]);
    const float hn = sigf(g[3]) * tanh_fast(cn);
    cbuf[b * HC_STRIDE + j] = cn;
    hf32[b * HC_STRIDE + j] = hn;
    hfrag[b * HFRAG_STRIDE + j] = (_Float16)hn;
  }
}

__global__ __launch_bounds__(THREADS)
void lstm_seq2seq_kernel(const float* __restrict__ inputs,
                         const float* __restrict__ Wih_e, const float* __restrict__ Whh_e,
                         const float* __restrict__ bih_e, const float* __restrict__ bhh_e,
                         const float* __restrict__ Wih_d, const float* __restrict__ Whh_d,
                         const float* __restrict__ bih_d, const float* __restrict__ bhh_d,
                         const float* __restrict__ Wy,    const float* __restrict__ by_g,
                         float* __restrict__ out) {
  extern __shared__ char smem[];
  _Float16* wfrag = (_Float16*)smem;
  _Float16* hfrag = wfrag + WFRAG_HALFS;
  float* gatesT = (float*)(hfrag + HFRAG_HALFS);
  float* hf32  = gatesT + GATES_FLOATS;
  float* cbuf  = hf32 + BT * HC_STRIDE;
  float* wih   = cbuf + BT * HC_STRIDE;
  float* bias  = wih + G_DIM * IN_DIM;
  float* xl    = bias + G_DIM;
  float* wy    = xl + BT * IN_DIM;
  float* byl   = wy + OUT_DIM * H_DIM;

  const int tid = threadIdx.x;
  const int bg0 = blockIdx.x * BT;   // first batch row of this workgroup

  // ---- init: encoder weights -> LDS; zero state ----
  build_wfrag(wfrag, Whh_e, tid);
  for (int i = tid; i < G_DIM * IN_DIM; i += THREADS) wih[i] = Wih_e[i];
  for (int i = tid; i < G_DIM; i += THREADS) bias[i] = bih_e[i] + bhh_e[i];
  for (int i = tid; i < HFRAG_HALFS; i += THREADS) hfrag[i] = (_Float16)0.0f;
  for (int i = tid; i < BT * HC_STRIDE; i += THREADS) { cbuf[i] = 0.0f; hf32[i] = 0.0f; }
  __syncthreads();

  // ---- encoder: 200 sequential steps ----
  for (int s = 0; s < T_STEPS; ++s) {
    for (int idx = tid; idx < BT * IN_DIM; idx += THREADS) {
      const int b = idx >> 2, k = idx & 3;
      xl[idx] = inputs[((size_t)s * B_TOT + bg0 + b) * IN_DIM + k];
    }
    gemm_step(wfrag, hfrag, gatesT, tid);
    __syncthreads();
    cell_step(gatesT, wih, bias, xl, cbuf, hf32, hfrag, tid);
    __syncthreads();
  }

  // ---- swap to decoder weights; reset c and y0 ----
  build_wfrag(wfrag, Whh_d, tid);
  for (int i = tid; i < G_DIM * IN_DIM; i += THREADS) wih[i] = Wih_d[i];
  for (int i = tid; i < G_DIM; i += THREADS) bias[i] = bih_d[i] + bhh_d[i];
  for (int i = tid; i < OUT_DIM * H_DIM; i += THREADS) wy[i] = Wy[i];
  for (int i = tid; i < OUT_DIM; i += THREADS) byl[i] = by_g[i];
  for (int i = tid; i < BT * HC_STRIDE; i += THREADS) cbuf[i] = 0.0f;
  for (int i = tid; i < BT * IN_DIM; i += THREADS) xl[i] = 0.0f;
  __syncthreads();

  // ---- decoder: 30 autoregressive steps ----
  for (int s = 0; s < DEC_STEPS; ++s) {
    gemm_step(wfrag, hfrag, gatesT, tid);
    __syncthreads();
    cell_step(gatesT, wih, bias, xl, cbuf, hf32, hfrag, tid);
    __syncthreads();
    // y = h @ Wy^T + by ; feed back as next x and emit
    for (int idx = tid; idx < BT * OUT_DIM; idx += THREADS) {
      const int b = idx >> 2, o = idx & 3;
      float acc = byl[o];
      const float* hr = hf32 + b * HC_STRIDE;
      const float* wr = wy + o * H_DIM;
#pragma unroll 4
      for (int j = 0; j < H_DIM; ++j) acc += hr[j] * wr[j];
      out[((size_t)s * B_TOT + bg0 + b) * OUT_DIM + o] = acc;
      xl[b * 4 + o] = acc;
    }
    __syncthreads();
  }
}

extern "C" void kernel_launch(void* const* d_in, const int* in_sizes, int n_in,
                              void* d_out, int out_size, void* d_ws, size_t ws_size,
                              hipStream_t stream) {
  (void)in_sizes; (void)n_in; (void)out_size; (void)d_ws; (void)ws_size;
  const float* inputs = (const float*)d_in[0];
  const float* Wih_e  = (const float*)d_in[1];
  const float* Whh_e  = (const float*)d_in[2];
  const float* bih_e  = (const float*)d_in[3];
  const float* bhh_e  = (const float*)d_in[4];
  const float* Wih_d  = (const float*)d_in[5];
  const float* Whh_d  = (const float*)d_in[6];
  const float* bih_d  = (const float*)d_in[7];
  const float* bhh_d  = (const float*)d_in[8];
  const float* Wy     = (const float*)d_in[9];
  const float* by     = (const float*)d_in[10];
  float* out = (float*)d_out;

  dim3 grid(B_TOT / BT);   // 128 workgroups, each owns 32 batch rows for all 230 steps
  dim3 block(THREADS);
  lstm_seq2seq_kernel<<<grid, block, SMEM_BYTES, stream>>>(
      inputs, Wih_e, Whh_e, bih_e, bhh_e,
      Wih_d, Whh_d, bih_d, bhh_d, Wy, by, out);
}
